// Attention_25194278159050
// MI455X (gfx1250) — matrix operand order repaired and checked by reading, not compiled, not run on hardware
//
#include <hip/hip_runtime.h>
#include <stdint.h>

// Problem constants (from reference): B=2, S=2048, D=768, H=12, Dh=64
#define S_LEN   2048
#define BATCH   2
#define DMODEL  768
#define NHEAD   12
#define DHEAD   64
#define NROWS   (BATCH * S_LEN)   // 4096 flattened (b,s) rows
#define IGNORE_VAL (-100000.0f)
#define P_STRIDE 40               // LDS P-tile row stride (16B-aligned b128 rows)

typedef __attribute__((ext_vector_type(16))) __bf16   v16bf;
typedef __attribute__((ext_vector_type(8)))  float    v8f;
typedef __attribute__((ext_vector_type(4)))  uint32_t v4u;

union BFrag { v16bf v; uint32_t u[8]; v4u q[2]; };

__device__ __forceinline__ uint16_t f2bf(float f) {
  uint32_t u = __builtin_bit_cast(uint32_t, f);
  u += 0x7FFFu + ((u >> 16) & 1u);          // round-to-nearest-even
  return (uint16_t)(u >> 16);
}
__device__ __forceinline__ uint32_t pack2(float lo, float hi) {
#if __has_builtin(__builtin_amdgcn_cvt_pk_bf16_f32)
  auto p = __builtin_amdgcn_cvt_pk_bf16_f32(lo, hi);   // v_cvt_pk_bf16_f32
  return __builtin_bit_cast(uint32_t, p);
#else
  return (uint32_t)f2bf(lo) | ((uint32_t)f2bf(hi) << 16);
#endif
}
__device__ __forceinline__ v8f wmma_bf16(const BFrag& a, const BFrag& b, v8f c) {
  return __builtin_amdgcn_wmma_f32_16x16x32_bf16(false, a.v, false, b.v,
                                                 (short)0, c, false, false);
}
// Two contiguous 8xbf16 runs -> one A/B fragment (16B-aligned addresses).
__device__ __forceinline__ void load_frag2(BFrag& f, const uint16_t* p, int second_off) {
  f.q[0] = *(const v4u*)p;
  f.q[1] = *(const v4u*)(p + second_off);
}

// ---------------------------------------------------------------------------
// Kernel 0a: straight f32 -> bf16 pair conversion (for x).
// ---------------------------------------------------------------------------
__global__ __launch_bounds__(256) void cvt_kernel(const float* __restrict__ src,
                                                  uint16_t* __restrict__ dst, int n2) {
  int i = blockIdx.x * 256 + threadIdx.x;
  if (i < n2) ((uint32_t*)dst)[i] = pack2(src[2 * i], src[2 * i + 1]);
}

// ---------------------------------------------------------------------------
// Kernel 0b: transpose-convert  src[H][K][N] f32  ->  dst[H][N][K] bf16,
// packing K-pairs so GEMM B-fragments load with b128.
// ---------------------------------------------------------------------------
__global__ __launch_bounds__(256) void cvtT_kernel(const float* __restrict__ src,
                                                   uint16_t* __restrict__ dst,
                                                   int K, int N, int total_pairs) {
  int i = blockIdx.x * 256 + threadIdx.x;
  if (i >= total_pairs) return;
  int kp = i % (K / 2);
  int hn = i / (K / 2);
  int h  = hn / N;
  int n  = hn % N;
  int k  = 2 * kp;
  float lo = src[((size_t)h * K + k)     * N + n];
  float hi = src[((size_t)h * K + k + 1) * N + n];
  ((uint32_t*)dst)[(size_t)hn * (K / 2) + kp] = pack2(lo, hi);
}

// ---------------------------------------------------------------------------
// Kernel 1: QKV projection.  One wave: 16(row) x 64(Dh) tile for one head,
// Q,K,V together (x A-fragment reused 12x).  Q pre-scaled by 1/sqrt(Dh).
// Q,K -> bf16 [B,H,S,Dh];  V -> transposed bf16 [B,H,Dh,S].
// B-fragments batched 4-wide so loads overlap the WMMA chain.
// ---------------------------------------------------------------------------
__global__ __launch_bounds__(32) void qkv_proj_kernel(
    const uint16_t* __restrict__ xb,
    const uint16_t* __restrict__ WQt, const uint16_t* __restrict__ WKt,
    const uint16_t* __restrict__ WVt,
    const float* __restrict__ bQ, const float* __restrict__ bK,
    const float* __restrict__ bV,
    uint16_t* __restrict__ Qb, uint16_t* __restrict__ Kb,
    uint16_t* __restrict__ Vt)
{
  const int lane = threadIdx.x;
  const int m  = lane & 15;
  const int kh = lane >> 4;
  const int r0 = blockIdx.x * 16;
  const int h  = blockIdx.y;

  const uint16_t* WT[3] = { WQt, WKt, WVt };
  v8f vzero = {};
  v8f acc[3][4];
#pragma unroll
  for (int w = 0; w < 3; ++w)
#pragma unroll
    for (int nt = 0; nt < 4; ++nt) acc[w][nt] = vzero;

#pragma unroll 1
  for (int kk = 0; kk < DMODEL / 32; ++kk) {
    BFrag a;   // A: row = r0+m, K-runs at kk*32 + 16*kh + {0,8}
    load_frag2(a, xb + (size_t)(r0 + m) * DMODEL + kk * 32 + 16 * kh, 8);
#pragma unroll
    for (int w = 0; w < 3; ++w) {
      const uint16_t* wp = WT[w] + (size_t)h * DHEAD * DMODEL + kk * 32 + 16 * kh;
      BFrag bf[4];
#pragma unroll
      for (int nt = 0; nt < 4; ++nt)             // issue all 4 loads first
        load_frag2(bf[nt], wp + (size_t)(nt * 16 + m) * DMODEL, 8);
#pragma unroll
      for (int nt = 0; nt < 4; ++nt)             // then the WMMA chain
        acc[w][nt] = wmma_bf16(a, bf[nt], acc[w][nt]);
    }
  }

  // Epilogue: D layout -> lane holds col N=m, rows M=j+8*kh
  const float* Bias[3] = { bQ, bK, bV };
#pragma unroll
  for (int w = 0; w < 3; ++w) {
#pragma unroll
    for (int nt = 0; nt < 4; ++nt) {
      int e = nt * 16 + m;
      float bias = Bias[w][h * DHEAD + e];
#pragma unroll
      for (int j = 0; j < 8; ++j) {
        int r  = r0 + j + 8 * kh;             // flattened (b,s)
        int bb = r >> 11;
        int s  = r & (S_LEN - 1);
        float val = acc[w][nt][j] + bias;
        if (w == 0) {
          Qb[(((size_t)bb * NHEAD + h) * S_LEN + s) * DHEAD + e] =
              f2bf(val * 0.125f);             // fold 1/sqrt(64) into Q
        } else if (w == 1) {
          Kb[(((size_t)bb * NHEAD + h) * S_LEN + s) * DHEAD + e] = f2bf(val);
        } else {
          Vt[(((size_t)bb * NHEAD + h) * DHEAD + e) * S_LEN + s] = f2bf(val);
        }
      }
    }
  }
}

// ---------------------------------------------------------------------------
// Kernel 2: flash attention. One wave per 16 query rows of one (b,h).
// 32-token K chunks: 4 WMMA (QK^T) -> online softmax (shfl row reductions)
// -> P staged through LDS (D-layout -> A-layout, ds b128) -> 4 WMMA (P.V).
// Writes Z bf16 [B,S,H*Dh] = [4096,768].
// ---------------------------------------------------------------------------
__global__ __launch_bounds__(32) void attn_kernel(
    const uint16_t* __restrict__ Qb, const uint16_t* __restrict__ Kb,
    const uint16_t* __restrict__ Vt, uint16_t* __restrict__ Zb)
{
  __shared__ uint16_t sP[16 * P_STRIDE];

  const int lane = threadIdx.x;
  const int m  = lane & 15;
  const int kh = lane >> 4;
  const int q0 = blockIdx.x * 16;
  const int h  = blockIdx.y;
  const int b  = blockIdx.z;
  const size_t base = ((size_t)(b * NHEAD + h)) * S_LEN * DHEAD;

  // Preload Q A-fragments (Dh=64 -> two K=32 chunks), 2x b128 each.
  BFrag qf[2];
#pragma unroll
  for (int c = 0; c < 2; ++c)
    load_frag2(qf[c], Qb + base + (size_t)(q0 + m) * DHEAD + 32 * c + 16 * kh, 8);

  float mrun[8], lrun[8];
#pragma unroll
  for (int j = 0; j < 8; ++j) { mrun[j] = -3.0e38f; lrun[j] = 0.0f; }
  v8f vzero = {};
  v8f accO[4];
#pragma unroll
  for (int nt = 0; nt < 4; ++nt) accO[nt] = vzero;

  const int nchunks = ((q0 + 15) >> 5) + 1;   // causal: only k0 <= q0+15
#pragma unroll 1
  for (int ck = 0; ck < nchunks; ++ck) {
    const int k0 = ck * 32;

    if (ck + 1 < nchunks) {                    // prefetch next K/V chunk
      __builtin_prefetch(Kb + base + (size_t)(k0 + 32 + lane) * DHEAD, 0, 1);
      __builtin_prefetch(Vt + base + (size_t)(lane * 2) * S_LEN + k0 + 32, 0, 1);
    }

    // ---- scores: two 16x16 D tiles; batch all 4 K-fragment loads ----
    BFrag bk[2][2];
#pragma unroll
    for (int t = 0; t < 2; ++t) {
      int tok = k0 + t * 16 + m;              // B-matrix col N = token
#pragma unroll
      for (int c = 0; c < 2; ++c)             // B rows K = dh run at 32c+16kh
        load_frag2(bk[t][c], Kb + base + (size_t)tok * DHEAD + 32 * c + 16 * kh, 8);
    }
    v8f st[2]; st[0] = vzero; st[1] = vzero;
#pragma unroll
    for (int t = 0; t < 2; ++t)
#pragma unroll
      for (int c = 0; c < 2; ++c)
        st[t] = wmma_bf16(qf[c], bk[t][c], st[t]);

    // ---- causal mask (cndmask, no EXEC divergence) ----
#pragma unroll
    for (int t = 0; t < 2; ++t)
#pragma unroll
      for (int j = 0; j < 8; ++j) {
        int q = q0 + j + 8 * kh;
        int k = k0 + t * 16 + m;
        st[t][j] = (k > q) ? IGNORE_VAL : st[t][j];
      }

    // ---- online softmax: row stats via 16-lane butterfly reductions ----
    float alpha[8];
#pragma unroll
    for (int j = 0; j < 8; ++j) {
      float v = fmaxf(st[0][j], st[1][j]);
#pragma unroll
      for (int xm = 1; xm < 16; xm <<= 1) v = fmaxf(v, __shfl_xor(v, xm, 32));
      float mnew = fmaxf(mrun[j], v);
      alpha[j] = __expf(mrun[j] - mnew);
      mrun[j]  = mnew;
    }
#pragma unroll
    for (int j = 0; j < 8; ++j) {
      float p0 = __expf(st[0][j] - mrun[j]);
      float p1 = __expf(st[1][j] - mrun[j]);
      st[0][j] = p0; st[1][j] = p1;
      float rs = p0 + p1;
#pragma unroll
      for (int xm = 1; xm < 16; xm <<= 1) rs += __shfl_xor(rs, xm, 32);
      lrun[j] = lrun[j] * alpha[j] + rs;
    }
#pragma unroll
    for (int nt = 0; nt < 4; ++nt)
#pragma unroll
      for (int j = 0; j < 8; ++j) accO[nt][j] *= alpha[j];

    // ---- stage P: D-layout (row=j+8kh, col=lane) -> LDS -> A-layout ----
#pragma unroll
    for (int t = 0; t < 2; ++t)
#pragma unroll
      for (int j = 0; j < 8; ++j)
        sP[(j + 8 * kh) * P_STRIDE + t * 16 + m] = f2bf(st[t][j]);
    __syncthreads();
    BFrag pf;                                  // ds_load_b128 x2
    load_frag2(pf, sP + m * P_STRIDE + 16 * kh, 8);
    __syncthreads();

    // ---- O += P.V : batch 4 V^T fragment loads, then 4 WMMAs ----
    BFrag bv[4];
#pragma unroll
    for (int nt = 0; nt < 4; ++nt) {
      int dh = nt * 16 + m;                   // B col N = dh
      load_frag2(bv[nt], Vt + base + (size_t)dh * S_LEN + k0 + 16 * kh, 8);
    }
#pragma unroll
    for (int nt = 0; nt < 4; ++nt)
      accO[nt] = wmma_bf16(pf, bv[nt], accO[nt]);
  }

  // ---- epilogue: O / l, write Z bf16 [B,S,H*Dh] ----
#pragma unroll
  for (int nt = 0; nt < 4; ++nt) {
    int dh = nt * 16 + m;
#pragma unroll
    for (int j = 0; j < 8; ++j) {
      int q = q0 + j + 8 * kh;
      size_t r = (size_t)b * S_LEN + q;
      Zb[r * DMODEL + h * DHEAD + dh] = f2bf(accO[nt][j] / lrun[j]);
    }
  }
}

// ---------------------------------------------------------------------------
// Kernel 3: output projection. Z[4096,768](bf16) x W_O^T[768,768](bf16)
// summed over (h,e), plus b_O.  One wave per 16(row) x 64(col) tile.
// ---------------------------------------------------------------------------
__global__ __launch_bounds__(32) void out_proj_kernel(
    const uint16_t* __restrict__ Zb, const uint16_t* __restrict__ WOt,
    const float* __restrict__ bO, float* __restrict__ out)
{
  const int lane = threadIdx.x;
  const int m  = lane & 15;
  const int kh = lane >> 4;
  const int r0 = blockIdx.x * 16;
  const int n0 = blockIdx.y * 64;

  v8f vzero = {};
  v8f acc[4];
#pragma unroll
  for (int nt = 0; nt < 4; ++nt) acc[nt] = vzero;

#pragma unroll 1
  for (int kk = 0; kk < DMODEL / 32; ++kk) {
    BFrag a;
    load_frag2(a, Zb + (size_t)(r0 + m) * DMODEL + kk * 32 + 16 * kh, 8);
    BFrag bw[4];
#pragma unroll
    for (int nt = 0; nt < 4; ++nt) {           // issue all 4 loads first
      int d = n0 + nt * 16 + m;
      load_frag2(bw[nt], WOt + (size_t)d * DMODEL + kk * 32 + 16 * kh, 8);
    }
#pragma unroll
    for (int nt = 0; nt < 4; ++nt)
      acc[nt] = wmma_bf16(a, bw[nt], acc[nt]);
  }

#pragma unroll
  for (int nt = 0; nt < 4; ++nt) {
    int d = n0 + nt * 16 + m;
    float bias = bO[d];
#pragma unroll
    for (int j = 0; j < 8; ++j) {
      int r = r0 + j + 8 * kh;
      out[(size_t)r * DMODEL + d] = acc[nt][j] + bias;
    }
  }
}

// ---------------------------------------------------------------------------
extern "C" void kernel_launch(void* const* d_in, const int* in_sizes, int n_in,
                              void* d_out, int out_size, void* d_ws, size_t ws_size,
                              hipStream_t stream) {
  const float* x  = (const float*)d_in[0];
  const float* WQ = (const float*)d_in[1];
  const float* WK = (const float*)d_in[2];
  const float* WV = (const float*)d_in[3];
  const float* WO = (const float*)d_in[4];
  const float* bQ = (const float*)d_in[5];
  const float* bK = (const float*)d_in[6];
  const float* bV = (const float*)d_in[7];
  const float* bO = (const float*)d_in[8];
  float* out = (float*)d_out;

  // Workspace layout (all bf16 / uint16_t):
  //  Qb,Kb [B,H,S,Dh]; Vt [B,H,Dh,S]; Zb [B,S,H*Dh]; xb [4096,768];
  //  WQt,WKt,WVt [H,Dh,D]; WOt [D, H*Dh].   Total ~= 34.5 MB.
  const size_t per  = (size_t)BATCH * NHEAD * S_LEN * DHEAD; // 3,145,728
  const size_t wsz  = (size_t)NHEAD * DHEAD * DMODEL;        //   589,824
  uint16_t* Qb  = (uint16_t*)d_ws;
  uint16_t* Kb  = Qb  + per;
  uint16_t* Vt  = Kb  + per;
  uint16_t* Zb  = Vt  + per;
  uint16_t* xb  = Zb  + per;          // NROWS*DMODEL == per
  uint16_t* WQt = xb  + per;
  uint16_t* WKt = WQt + wsz;
  uint16_t* WVt = WKt + wsz;
  uint16_t* WOt = WVt + wsz;

  // --- stage 0: one-time conversions / transposes to bf16 ---
  {
    int n2 = (int)(per / 2);                               // x pairs
    cvt_kernel<<<(n2 + 255) / 256, 256, 0, stream>>>(x, xb, n2);
    int tp = (int)(wsz / 2);                               // weight pairs
    cvtT_kernel<<<(tp + 255) / 256, 256, 0, stream>>>(WQ, WQt, DMODEL, DHEAD, tp);
    cvtT_kernel<<<(tp + 255) / 256, 256, 0, stream>>>(WK, WKt, DMODEL, DHEAD, tp);
    cvtT_kernel<<<(tp + 255) / 256, 256, 0, stream>>>(WV, WVt, DMODEL, DHEAD, tp);
    cvtT_kernel<<<(tp + 255) / 256, 256, 0, stream>>>(WO, WOt, DMODEL, DMODEL, tp);
  }

  dim3 g1(NROWS / 16, NHEAD);
  qkv_proj_kernel<<<g1, 32, 0, stream>>>(xb, WQt, WKt, WVt, bQ, bK, bV, Qb, Kb, Vt);

  dim3 g2(S_LEN / 16, NHEAD, BATCH);
  attn_kernel<<<g2, 32, 0, stream>>>(Qb, Kb, Vt, Zb);

  dim3 g3(NROWS / 16, DMODEL / 64);
  out_proj_kernel<<<g3, 32, 0, stream>>>(Zb, WOt, bO, out);
}
